// AttentionDot_64123861729626
// MI455X (gfx1250) — matrix operand order repaired and checked by reading, not compiled, not run on hardware
//
#include <hip/hip_runtime.h>
#include <math.h>

#define BB 8
#define TQ 256
#define TP 256
#define DD 128
#define APAD 136   // padded LDS row stride in halves (68 dwords -> conflict-free b128 loads)

typedef __attribute__((ext_vector_type(16))) _Float16 v16h;
typedef __attribute__((ext_vector_type(8)))  _Float16 v8h;
typedef __attribute__((ext_vector_type(8)))  float    v8f;

// gfx1250 hardware tanh (trans op); branch-free fallback otherwise.
__device__ __forceinline__ float fast_tanh(float x) {
#if __has_builtin(__builtin_amdgcn_tanhf)
    return __builtin_amdgcn_tanhf(x);
#else
    const float e = __builtin_amdgcn_exp2f(x * 2.885390082f);   // exp(2x)
    return 1.0f - 2.0f * __builtin_amdgcn_rcpf(e + 1.0f);
#endif
}

// ---------------------------------------------------------------------------
// Kernel 0: precision/layout prep.
//   Wt[e][d]    = (f16) W[d][e]       (WMMA A-fragment source for scores GEMM)
//   qT[b][d][t] = (f16) q[b][t][d]    (WMMA B-fragment source for output GEMM)
// ---------------------------------------------------------------------------
__global__ __launch_bounds__(256) void prep_kernel(
    const float* __restrict__ q, const float* __restrict__ W,
    _Float16* __restrict__ Wt, _Float16* __restrict__ qT)
{
    const int idx = blockIdx.x * 256 + threadIdx.x;
    if (idx < DD * DD) {
        const int d = idx / DD, e = idx % DD;          // read W coalesced
        Wt[e * DD + d] = (_Float16)W[idx];
    }
    if (idx < BB * TQ * DD) {
        const int d = idx % DD;
        const int t = (idx / DD) % TQ;
        const int b = idx / (DD * TQ);                 // read q coalesced
        qT[((size_t)b * DD + d) * TQ + t] = (_Float16)q[idx];
    }
}

// ---------------------------------------------------------------------------
// Kernel 1: scores[b][p][t] = sum_e tanh( (q[b,t,:]*p[b,p,:]) @ W )[e] * vd[e]
//
// Transposed GEMM: C(m=e, n=t) = sum_d Wt[e][d] * A[t][d].
//   A-operand (M=e, K=d)  = Wt rows  -> per-wave constant, loaded once.
//   B-operand (K=d, N=t)  = q*p tile -> contiguous per-lane LDS loads.
// Lane n owns a t column => e-reduction is per-lane FMA + one half-swap shfl.
// One workgroup (8 waves) per (b,p). Wave w owns e-rows [16w, 16w+16).
// ---------------------------------------------------------------------------
__global__ __launch_bounds__(256) void scores_kernel(
    const float* __restrict__ q, const float* __restrict__ p,
    const float* __restrict__ vd, const _Float16* __restrict__ Wt,
    float* __restrict__ scores)
{
    __shared__ _Float16 A[128][APAD];   // half of the 256x128 q*p tile, f16
    __shared__ float    ps[DD];
    __shared__ float    sc[TQ];

    const int b    = blockIdx.x / TP;
    const int pp   = blockIdx.x % TP;
    const int tid  = threadIdx.x;
    const int lane = tid & 31;
    const int wave = tid >> 5;
    const int l15  = lane & 15;
    const int hi   = lane >> 4;         // 0: lanes 0-15, 1: lanes 16-31

    if (tid < DD)  ps[tid] = p[((size_t)b * TP + pp) * DD + tid];
    if (tid < TQ)  sc[tid] = 0.0f;

    // hint the (tiny, L2-resident) Wt matrix toward the WGP
    __builtin_prefetch(Wt + (size_t)wave * 16 * DD, 0, 0);

    // Per-wave A fragments (M=e,K=d), contiguous k order, same as B:
// lane m = l15 holds row e = 16*wave + m; lanes 0-15 K=0..15,
// lanes 16-31 K=16..31 within each 32-wide K chunk.
    v16h afr[4];
    {
        const _Float16* ar = Wt + (size_t)(wave * 16 + l15) * DD + hi * 16;
        #pragma unroll
        for (int kc = 0; kc < 4; ++kc) {
            v8h a0 = *(const v8h*)(ar + kc * 32);
            v8h a1 = *(const v8h*)(ar + kc * 32 + 8);
            afr[kc] = __builtin_shufflevector(a0, a1,
                      0,1,2,3,4,5,6,7,8,9,10,11,12,13,14,15);
        }
    }
    // vd values for this lane's 8 e-rows: e = 16*wave + 8*hi + v
    const v8f vdv = *(const v8f*)(vd + wave * 16 + hi * 8);

    __syncthreads();

    for (int pass = 0; pass < 2; ++pass) {
        if (pass) __syncthreads();      // all reads of previous A complete
        // Build A rows [128*pass, 128*pass+128): thread -> (row, half-row)
        {
            const int r     = tid >> 1;
            const int dbase = (tid & 1) * 64;
            const float* qr = q + ((size_t)b * TQ + pass * 128 + r) * DD + dbase;
            #pragma unroll
            for (int j = 0; j < 64; j += 4) {
                float4 qv = *(const float4*)(qr + j);
                A[r][dbase + j + 0] = (_Float16)(qv.x * ps[dbase + j + 0]);
                A[r][dbase + j + 1] = (_Float16)(qv.y * ps[dbase + j + 1]);
                A[r][dbase + j + 2] = (_Float16)(qv.z * ps[dbase + j + 2]);
                A[r][dbase + j + 3] = (_Float16)(qv.w * ps[dbase + j + 3]);
            }
        }
        __syncthreads();

        // 8 t-tiles of 16 in this pass; wave covers all of them for its e-rows
        for (int rt = 0; rt < 8; ++rt) {
            // B fragments (K=d, N=t): lane n = l15 -> column t = rt*16 + n,
            // lanes 0-15 K=0..15, lanes 16-31 K=16..31: contiguous 32B loads.
            const _Float16* bp = &A[rt * 16 + l15][hi * 16];
            v8f c = {};
            #pragma unroll
            for (int kc = 0; kc < 4; ++kc) {
                v16h bfr = *(const v16h*)(bp + kc * 32);
                c = __builtin_amdgcn_wmma_f32_16x16x32_f16(
                        false, afr[kc], false, bfr, (short)0, c, false, false);
            }
            // C element (m = v + 8*hi, n = l15): lane owns a t column.
            // Per-lane e-reduction, then fold the two half-wave m-halves.
            float s = 0.0f;
            #pragma unroll
            for (int v = 0; v < 8; ++v)
                s = fmaf(fast_tanh(c[v]), vdv[v], s);
            s += __shfl_xor(s, 16, 32);
            if (hi == 0)
                atomicAdd(&sc[pass * 128 + rt * 16 + l15], s);
        }
    }
    __syncthreads();
    if (tid < TQ)
        scores[((size_t)b * TP + pp) * TQ + tid] = sc[tid];
}

// ---------------------------------------------------------------------------
// Kernel 2: softmax over the p axis (column softmax of [p][t] per batch),
// emit f16 weights in [b][p][t] layout (row-major over t).
// ---------------------------------------------------------------------------
__global__ __launch_bounds__(256) void softmax_kernel(
    const float* __restrict__ scores, _Float16* __restrict__ wgt)
{
    const int b = blockIdx.x;
    const int t = threadIdx.x;
    const float* col = scores + (size_t)b * TP * TQ + t;
    float m = -INFINITY;
    for (int pp = 0; pp < TP; ++pp) m = fmaxf(m, col[(size_t)pp * TQ]);
    float s = 0.0f;
    for (int pp = 0; pp < TP; ++pp) s += __expf(col[(size_t)pp * TQ] - m);
    const float inv = 1.0f / s;
    _Float16* wcol = wgt + (size_t)b * TP * TQ + t;
    for (int pp = 0; pp < TP; ++pp)
        wcol[(size_t)pp * TQ] = (_Float16)(__expf(col[(size_t)pp * TQ] - m) * inv);
}

// ---------------------------------------------------------------------------
// Kernel 3: out[b][p][d] = sum_t w[b][p][t] * q[b][t][d]
// WG per (b, 16-row p-tile); wave w owns d-columns [16w,16w+16); K = 256.
// ---------------------------------------------------------------------------
__global__ __launch_bounds__(256) void outgemm_kernel(
    const _Float16* __restrict__ wgt, const _Float16* __restrict__ qT,
    float* __restrict__ out)
{
    const int b    = blockIdx.x >> 4;
    const int rt   = blockIdx.x & 15;
    const int lane = threadIdx.x & 31;
    const int wave = threadIdx.x >> 5;
    const int l15  = lane & 15;
    const int hi   = lane >> 4;

    const _Float16* arow = wgt + ((size_t)b * TP + rt * 16 + l15) * TQ + hi * 16;
    const _Float16* brow = qT  + ((size_t)b * DD + wave * 16 + l15) * TQ + hi * 16;

    v8f c = {};
    #pragma unroll
    for (int kc = 0; kc < 8; ++kc) {
        v8h a0 = *(const v8h*)(arow + kc * 32);
        v8h a1 = *(const v8h*)(arow + kc * 32 + 8);
        v16h a = __builtin_shufflevector(a0, a1,
                 0,1,2,3,4,5,6,7,8,9,10,11,12,13,14,15);
        v16h bfrag = *(const v16h*)(brow + kc * 32);
        c = __builtin_amdgcn_wmma_f32_16x16x32_f16(
                false, a, false, bfrag, (short)0, c, false, false);
    }
    #pragma unroll
    for (int v = 0; v < 8; ++v) {
        const int m = v + (hi << 3);
        out[((size_t)b * TP + rt * 16 + m) * DD + wave * 16 + l15] = c[v];
    }
}

// ---------------------------------------------------------------------------
extern "C" void kernel_launch(void* const* d_in, const int* in_sizes, int n_in,
                              void* d_out, int out_size, void* d_ws, size_t ws_size,
                              hipStream_t stream) {
    const float* q  = (const float*)d_in[0];   // [8,256,128]
    const float* p  = (const float*)d_in[1];   // [8,256,128]
    const float* W  = (const float*)d_in[2];   // [128,128]
    const float* vd = (const float*)d_in[3];   // [128,1]
    float* out = (float*)d_out;                // [8,256,128]

    // Workspace layout (bytes):
    //   [0, 2MB)            : scores f32 [B][TP][TQ]
    //   [2MB, 3MB)          : weights f16 [B][TP][TQ]
    //   [3MB, 3MB+64KB)     : Wt f16 [D][D]
    //   [3MB+64KB, +512KB)  : qT f16 [B][D][TQ]
    char* ws = (char*)d_ws;
    float*    scores = (float*)ws;
    _Float16* wgt    = (_Float16*)(ws + (size_t)2 * 1024 * 1024);
    _Float16* Wt     = (_Float16*)(ws + (size_t)3 * 1024 * 1024);
    _Float16* qT     = (_Float16*)(ws + (size_t)3 * 1024 * 1024 + 64 * 1024);

    (void)in_sizes; (void)n_in; (void)out_size; (void)ws_size;

    prep_kernel   <<<(BB * TQ * DD + 255) / 256, 256, 0, stream>>>(q, W, Wt, qT);
    scores_kernel <<<BB * TP,                    256, 0, stream>>>(q, p, vd, Wt, scores);
    softmax_kernel<<<BB,                         256, 0, stream>>>(scores, wgt);
    outgemm_kernel<<<BB * 16,                    256, 0, stream>>>(wgt, qT, out);
}
